// SGGN_layer_33062658245058
// MI455X (gfx1250) — matrix operand order repaired and checked by reading, not compiled, not run on hardware
//
#include <hip/hip_runtime.h>
#include <hip/hip_bf16.h>
#include <math.h>

typedef __attribute__((ext_vector_type(16))) _Float16 v16h;
typedef __attribute__((ext_vector_type(8)))  _Float16 v8h;
typedef __attribute__((ext_vector_type(8)))  float    v8f;

#define NN     8192
#define DD     64
#define MAXDEG 16
#define KE     8
#define LL     17      // MAXDEG + 1
#define LP     32      // padded L for two 16-row WMMA M-tiles
#define DIN    128
#define TDIN   256
#define EPS_   1e-5f

__device__ __forceinline__ float sigmoidf_(float x) { return 1.0f / (1.0f + __expf(-x)); }
__device__ __forceinline__ float siluf_(float x)    { return x / (1.0f + __expf(-x)); }

// ---------------------------------------------------------------------------
// K0: convert GEMM weights to f16 (B-operand, row-major = [out_col][k]) and
//     zero the BatchNorm partial-sum accumulators (must be re-zeroed per call).
// ---------------------------------------------------------------------------
__global__ void k0_prep(const float* __restrict__ fc1w, const float* __restrict__ fc2w,
                        _Float16* __restrict__ fc1h, _Float16* __restrict__ fc2h,
                        float* __restrict__ bnAcc) {
    int i = blockIdx.x * blockDim.x + threadIdx.x;
    if (i < TDIN * DD) fc1h[i] = (_Float16)fc1w[i];   // [256][64]
    if (i < DD * DIN)  fc2h[i] = (_Float16)fc2w[i];   // [64][128]
    if (i < 128)       bnAcc[i] = 0.0f;               // bnSum[64] + bnSumSq[64]
}

// ---------------------------------------------------------------------------
// K1: messages + noise-rank "mailbox sort" -> hflat [N, 16*64] (slots 8..15 = 0)
//     dst[e] == e/8, so node n owns edges [8n, 8n+8); score order == noise order.
// ---------------------------------------------------------------------------
__global__ void k1_messages(const float* __restrict__ hs, const float* __restrict__ hs_e,
                            const float* __restrict__ noise, float* __restrict__ hflat) {
    int n = blockIdx.x;
    int tid = threadIdx.x;                 // 64 threads: one per feature
    __shared__ float sc[KE];
    __shared__ int   rk[KE];
    if (tid < KE) sc[tid] = noise[n * KE + tid];
    __syncthreads();
    if (tid < KE) {
        float s = sc[tid]; int r = 0;
        #pragma unroll
        for (int j = 0; j < KE; ++j) {
            float o = sc[j];
            r += (o < s) || (o == s && j < tid);   // stable argsort rank
        }
        rk[tid] = r;
    }
    __syncthreads();
    float hval = hs[n * DD + tid];
    float* base = hflat + (size_t)n * (MAXDEG * DD);
    #pragma unroll
    for (int j = 0; j < KE; ++j) {
        float m = hval * sigmoidf_(hs_e[(size_t)(n * KE + j) * DD + tid]);
        base[rk[j] * DD + tid] = m;
    }
    #pragma unroll
    for (int s = KE; s < MAXDEG; ++s) base[s * DD + tid] = 0.0f;
}

// ---------------------------------------------------------------------------
// K3: fused (A+I)-SPMM + GatedCNN per node via v_wmma_f32_16x16x32_f16.
//     128 threads = 4 waves. h_input is built directly in LDS (never spilled
//     to global): saves the 71 MB h_input round trip of the unfused version.
// ---------------------------------------------------------------------------
__global__ __launch_bounds__(128) void k3_spmm_gatedcnn(
    const float* __restrict__ hs, const int* __restrict__ src,
    const float* __restrict__ hflat,
    const _Float16* __restrict__ fc1h, const float* __restrict__ fc1b,
    const float* __restrict__ convw, const float* __restrict__ convb,
    const _Float16* __restrict__ fc2h, const float* __restrict__ fc2b,
    const float* __restrict__ rmsw,
    float* __restrict__ Y, float* __restrict__ bnSum, float* __restrict__ bnSumSq) {

    int n    = blockIdx.x;
    int tid  = threadIdx.x;
    int wv   = tid >> 5;
    int lane = tid & 31;
    int lh   = lane & 15;
    int hi16 = (lane >> 4) & 1;

    __shared__ int      s8[KE];
    __shared__ float    sHin[LL * DD];   // h_input f32 (for residual)        (4.25 KB)
    __shared__ _Float16 sA[LP * DD];     // h_input f16, rows 17..31 zero     (4 KB)
    __shared__ float    sXZ[LP * TDIN];  // fc1 output                        (32 KB)
    __shared__ _Float16 sG[LP * DIN];    // gated activations f16             (8 KB)
    __shared__ float    sO[LP * DD];     // fc2 output                        (8 KB)
    __shared__ float    sR[LP];          // rms factors

    if (tid < KE) {
        int sj = src[n * KE + tid];
        s8[tid] = sj;
        // warm the gathered mailbox rows (global_prefetch_b8); 4 lines apart
        const float* rowp = hflat + (size_t)sj * (MAXDEG * DD);
        __builtin_prefetch(rowp, 0, 0);
        __builtin_prefetch(rowp + 256, 0, 0);
        __builtin_prefetch(rowp + 512, 0, 0);
        __builtin_prefetch(rowp + 768, 0, 0);
    }
    __syncthreads();

    // ---- fused SPMM: h_input row 0 = hs[n]; rows 1..16 = self + 8 gathers --
    const float* self = hflat + (size_t)n * (MAXDEG * DD);
    for (int i = tid; i < LL * DD; i += 128) {
        float val;
        if (i < DD) {
            val = hs[n * DD + i];
        } else {
            int ii = i - DD;
            val = self[ii];
            #pragma unroll
            for (int j = 0; j < KE; ++j)
                val += hflat[(size_t)s8[j] * (MAXDEG * DD) + ii];
        }
        sHin[i] = val;
        sA[i]   = (_Float16)val;
    }
    for (int i = LL * DD + tid; i < LP * DD; i += 128) sA[i] = (_Float16)0.0f;
    __syncthreads();

    // ---- fc1: [32x64] x [64x256] ------------------------------------------
    // hoist A fragments: (mtile, kstep) = 2x2; contiguous k = 16*hi16 + s order (same as B)
    v16h afrag[2][2];
    #pragma unroll
    for (int mt = 0; mt < 2; ++mt)
        #pragma unroll
        for (int kk = 0; kk < 2; ++kk) {
            int m = mt * 16 + lh;
            const _Float16* p = sA + m * DD + kk * 32;
            v8h lo = *(const v8h*)(p + (hi16 ? 16 : 0));
            v8h hh = *(const v8h*)(p + (hi16 ? 24 : 8));
            v16h a;
            #pragma unroll
            for (int q = 0; q < 8; ++q) { a[q] = lo[q]; a[8 + q] = hh[q]; }
            afrag[mt][kk] = a;
        }

    #pragma unroll
    for (int j = 0; j < 4; ++j) {                  // 4 N-tiles per wave
        int ncol = (wv * 4 + j) * 16 + lh;
        float bias = fc1b[ncol];
        #pragma unroll
        for (int mt = 0; mt < 2; ++mt) {
            v8f acc = {};
            #pragma unroll
            for (int kk = 0; kk < 2; ++kk) {
                // B 32x16: lanes 0-15 hold K=0..15, lanes 16-31 hold K=16..31
                v16h b = *(const v16h*)(fc1h + ncol * DD + kk * 32 + (hi16 ? 16 : 0));
                acc = __builtin_amdgcn_wmma_f32_16x16x32_f16(
                          false, afrag[mt][kk], false, b, (short)0, acc, false, false);
            }
            int mbase = mt * 16 + (hi16 ? 8 : 0);  // D layout: M = vgpr + 8*hi16
            #pragma unroll
            for (int q = 0; q < 8; ++q)
                sXZ[(mbase + q) * TDIN + ncol] = acc[q] + bias;
        }
    }
    __syncthreads();

    // ---- causal depthwise conv (DCONV=4) + SiLU gating --------------------
    {
        int c = tid;                                // channel 0..127
        float w0 = convw[c * 4 + 0], w1 = convw[c * 4 + 1];
        float w2 = convw[c * 4 + 2], w3 = convw[c * 4 + 3];
        float cb = convb[c];
        for (int m = 0; m < LL; ++m) {
            float acc = cb;
            if (m >= 3) acc += w0 * sXZ[(m - 3) * TDIN + c];
            if (m >= 2) acc += w1 * sXZ[(m - 2) * TDIN + c];
            if (m >= 1) acc += w2 * sXZ[(m - 1) * TDIN + c];
            acc += w3 * sXZ[m * TDIN + c];
            float z = sXZ[m * TDIN + DIN + c];
            sG[m * DIN + c] = (_Float16)(siluf_(acc) * siluf_(z));
        }
        for (int m = LL; m < LP; ++m) sG[m * DIN + c] = (_Float16)0.0f;
    }
    __syncthreads();

    // ---- fc2: [32x128] x [128x64] -----------------------------------------
    {
        int ncol = wv * 16 + lh;                    // wave -> one 16-col tile of 64
        float bias = fc2b[ncol];
        #pragma unroll
        for (int mt = 0; mt < 2; ++mt) {
            v8f acc = {};
            #pragma unroll
            for (int kk = 0; kk < 4; ++kk) {
                int m = mt * 16 + lh;
                const _Float16* p = sG + m * DIN + kk * 32;
                v8h lo = *(const v8h*)(p + (hi16 ? 16 : 0));
                v8h hh = *(const v8h*)(p + (hi16 ? 24 : 8));
                v16h a;
                #pragma unroll
                for (int q = 0; q < 8; ++q) { a[q] = lo[q]; a[8 + q] = hh[q]; }
                v16h b = *(const v16h*)(fc2h + ncol * DIN + kk * 32 + (hi16 ? 16 : 0));
                acc = __builtin_amdgcn_wmma_f32_16x16x32_f16(
                          false, a, false, b, (short)0, acc, false, false);
            }
            int mbase = mt * 16 + (hi16 ? 8 : 0);
            #pragma unroll
            for (int q = 0; q < 8; ++q)
                sO[(mbase + q) * DD + ncol] = acc[q] + bias;
        }
    }
    __syncthreads();

    // ---- RMSNorm factors over D -------------------------------------------
    if (tid < LL) {
        float ss = 0.0f;
        for (int j = 0; j < DD; ++j) { float v = sO[tid * DD + j]; ss += v * v; }
        sR[tid] = rsqrtf(ss * (1.0f / DD) + EPS_);
    }
    __syncthreads();

    // ---- y = rmsnorm(out)*rms_w + h_input; write Y; BN partial sums -------
    if (tid < DD) {
        int j = tid;
        float rw = rmsw[j];
        float s = 0.0f, s2 = 0.0f;
        float* yrow = Y + (size_t)n * (LL * DD);
        for (int m = 0; m < LL; ++m) {
            float y = sO[m * DD + j] * sR[m] * rw + sHin[m * DD + j];
            yrow[m * DD + j] = y;
            s += y; s2 += y * y;
        }
        atomicAdd(&bnSum[j], s);
        atomicAdd(&bnSumSq[j], s2);
    }
}

// ---------------------------------------------------------------------------
// K4: BatchNorm (training stats) + ReLU6 + Linear(L,1) agg + residual
// ---------------------------------------------------------------------------
__global__ void k4_bn_agg(const float* __restrict__ Y, const float* __restrict__ bnSum,
                          const float* __restrict__ bnSumSq, const float* __restrict__ gamma,
                          const float* __restrict__ beta, const float* __restrict__ aggw,
                          const float* __restrict__ aggb, const float* __restrict__ hs,
                          float* __restrict__ out) {
    int idx = blockIdx.x * blockDim.x + threadIdx.x;   // over N*D
    int d = idx & (DD - 1);
    int n = idx >> 6;
    const float invCnt = 1.0f / (float)(NN * LL);
    float mean  = bnSum[d] * invCnt;
    float var   = bnSumSq[d] * invCnt - mean * mean;   // biased, as torch BN training
    float scale = gamma[d] * rsqrtf(var + EPS_);
    float shift = beta[d] - scale * mean;
    const float* yrow = Y + (size_t)n * (LL * DD) + d;
    float acc = 0.0f;
    #pragma unroll
    for (int m = 0; m < LL; ++m) {
        float yb = scale * yrow[m * DD] + shift;
        yb = fminf(fmaxf(yb, 0.0f), 6.0f);
        acc += yb * aggw[m];
    }
    out[idx] = acc + aggb[0] + hs[idx];
}

// ---------------------------------------------------------------------------
extern "C" void kernel_launch(void* const* d_in, const int* in_sizes, int n_in,
                              void* d_out, int out_size, void* d_ws, size_t ws_size,
                              hipStream_t stream) {
    const float* hs     = (const float*)d_in[0];
    const float* hs_e   = (const float*)d_in[1];
    // d_in[2] = degree (constant K, does not affect the sort -> unused)
    const float* noise  = (const float*)d_in[3];
    const float* fc1w   = (const float*)d_in[4];
    const float* fc1b   = (const float*)d_in[5];
    const float* convw  = (const float*)d_in[6];
    const float* convb  = (const float*)d_in[7];
    const float* fc2w   = (const float*)d_in[8];
    const float* fc2b   = (const float*)d_in[9];
    const float* rmsw   = (const float*)d_in[10];
    const float* gamma  = (const float*)d_in[11];
    const float* beta   = (const float*)d_in[12];
    const float* aggw   = (const float*)d_in[13];
    const float* aggb   = (const float*)d_in[14];
    const int*   src    = (const int*)d_in[15];
    // d_in[16] = dst (== e/K by construction -> unused)
    float* out = (float*)d_out;

    // workspace layout (bytes): weights | bn acc | hflat (32 MB) | Y (35.7 MB)
    char* ws = (char*)d_ws;
    _Float16* fc1h  = (_Float16*)(ws + 0);           // 32 KB
    _Float16* fc2h  = (_Float16*)(ws + 32768);       // 16 KB
    float*    bnAcc = (float*)   (ws + 49152);       // bnSum[64] | bnSumSq[64]
    float*    bnSum = bnAcc;
    float*    bnSqS = bnAcc + 64;
    float*    hflat = (float*)   (ws + 65536);                              // [N,16*64]
    float*    Yb    = (float*)   (ws + 65536 + (size_t)NN*MAXDEG*DD*4);     // [N,17,64]

    k0_prep<<<(TDIN * DD + 255) / 256, 256, 0, stream>>>(fc1w, fc2w, fc1h, fc2h, bnAcc);
    k1_messages<<<NN, 64, 0, stream>>>(hs, hs_e, noise, hflat);
    k3_spmm_gatedcnn<<<NN, 128, 0, stream>>>(hs, src, hflat, fc1h, fc1b, convw, convb,
                                             fc2h, fc2b, rmsw, Yb, bnSum, bnSqS);
    k4_bn_agg<<<(NN * DD) / 256, 256, 0, stream>>>(Yb, bnSum, bnSqS, gamma, beta,
                                                   aggw, aggb, hs, out);
}
